// ConvAConnect_51866025066814
// MI455X (gfx1250) — compile-verified, matrix operand order checked by reading, not run on hardware
//
#include <hip/hip_runtime.h>

// Problem constants (from reference)
#define BATCH 64
#define HIN   64
#define WIN   64
#define CINCH 128
#define COUTC 256
#define HO    62
#define WO    62
#define NPIX  (HO * WO)                 // 3844 output pixels per sample
#define MTILE 64
#define NMT   ((NPIX + MTILE - 1) / MTILE)  // 61 M-tiles per sample
#define NKCH  36                        // 3*3*(128/32) K-chunks of 32

typedef __attribute__((ext_vector_type(16))) __bf16 v16bf;
typedef __attribute__((ext_vector_type(8)))  float  v8f;

// LDS tiles, padded row stride to avoid bank conflicts (80B = 20 dwords,
// 20*n mod 64 distinct for n=0..15). Double buffered (ping-pong).
#define A_STRIDE 80                     // 64 rows  x (32 bf16 + pad)
#define B_STRIDE 80                     // 256 cols x (32 bf16 + pad)
#define A_BYTES  (MTILE * A_STRIDE)     // 5120 B
#define B_BYTES  (COUTC * B_STRIDE)     // 20480 B

// fp32 -> bf16, round-to-nearest-even, returned in low 16 bits
__device__ __forceinline__ unsigned int f2bf(float f) {
  unsigned int u = __float_as_uint(f);
  return (u + 0x7fffu + ((u >> 16) & 1u)) >> 16;
}

struct Frag32 { uint4 lo, hi; };        // 32 bytes == v16bf

__global__ __launch_bounds__(256) void conv_aconnect_wmma(
    const float* __restrict__ X,     // [64,64,64,128]
    const float* __restrict__ W,     // [3,3,128,256]
    const float* __restrict__ bias,  // [256]
    const float* __restrict__ Werr,  // [64,3,3,128,256]
    const float* __restrict__ Berr,  // [64,256]
    float* __restrict__ out)         // [64,62,62,256]
{
  __shared__ __align__(16) unsigned char ldsA[2][A_BYTES];
  __shared__ __align__(16) unsigned char ldsB[2][B_BYTES];

  const int t    = threadIdx.x;
  const int b    = blockIdx.x / NMT;
  const int m0   = (blockIdx.x % NMT) * MTILE;

  const int lane = t & 31;
  const int wave = t >> 5;
  const int wm   = wave >> 1;          // 0..3 : 16-row band within M-tile
  const int wn   = wave & 1;           // 0..1 : 128-col band within N=256
  const int half = lane >> 4;          // K-half selector (ISA A/B layout)
  const int l16  = lane & 15;

  // A-staging mapping: 4 threads per pixel, 8 channels each
  const int ap = t >> 2;               // pixel in tile (0..63)
  const int aq = t & 3;                // channel octet (0..3)
  const int apix   = m0 + ap;
  const bool avalid = apix < NPIX;
  const int aoh = avalid ? (apix / WO) : 0;
  const int aow = avalid ? (apix % WO) : 0;

  // B-staging mapping: lane-varying k-pair, 16-col block per thread
  const int brp = t & 15;              // k-pair index -> k0 = 2*brp
  const int bc0 = (t >> 4) * 16;       // col block base (0..240)
  const int k0  = 2 * brp;

  v8f acc[8];
#pragma unroll
  for (int j = 0; j < 8; ++j)
    acc[j] = (v8f){0.f,0.f,0.f,0.f,0.f,0.f,0.f,0.f};

  const float* __restrict__ WerrB = Werr + (size_t)b * (9 * CINCH * COUTC);

  // ---- register prefetch state for the next K-chunk ----
  float4 px0, px1;                     // X octets (8 floats)
  float4 pw[8], pe[8];                 // W rows k0,k0+1 (32f) and Werr rows (32f)

  // Issue global loads for chunk (kh,kw,cc) into registers (no waits here).
  auto issue_loads = [&](int kh, int kw, int cc) {
    px0 = (float4){0,0,0,0};
    px1 = (float4){0,0,0,0};
    if (avalid) {
      const float* src = X + (((size_t)b * HIN + (aoh + kh)) * WIN
                               + (aow + kw)) * CINCH + cc + aq * 8;
      px0 = *(const float4*)(src);
      px1 = *(const float4*)(src + 4);
    }
    const size_t rowOff = ((size_t)(kh * 3 + kw) * CINCH + (cc + k0)) * COUTC + bc0;
    const float4* w0 = (const float4*)(W     + rowOff);
    const float4* w1 = (const float4*)(W     + rowOff + COUTC);
    const float4* e0 = (const float4*)(WerrB + rowOff);
    const float4* e1 = (const float4*)(WerrB + rowOff + COUTC);
#pragma unroll
    for (int v = 0; v < 4; ++v) {
      pw[v]     = w0[v];
      pw[4 + v] = w1[v];
      pe[v]     = e0[v];
      pe[4 + v] = e1[v];
    }
  };

  // Consume the register prefetch: multiply/convert and store into LDS buf.
  auto store_chunk = [&](int buf) {
    // A tile
    uint4 d;
    d.x = f2bf(px0.x) | (f2bf(px0.y) << 16);
    d.y = f2bf(px0.z) | (f2bf(px0.w) << 16);
    d.z = f2bf(px1.x) | (f2bf(px1.y) << 16);
    d.w = f2bf(px1.z) | (f2bf(px1.w) << 16);
    *(uint4*)(ldsA[buf] + ap * A_STRIDE + aq * 16) = d;
    // B tile: memW = W * Werr, packed as {k0, k0+1} bf16 pairs per column
    float r0[16], r1[16];
#pragma unroll
    for (int v = 0; v < 4; ++v) {
      r0[4*v+0] = pw[v].x * pe[v].x;       r0[4*v+1] = pw[v].y * pe[v].y;
      r0[4*v+2] = pw[v].z * pe[v].z;       r0[4*v+3] = pw[v].w * pe[v].w;
      r1[4*v+0] = pw[4+v].x * pe[4+v].x;   r1[4*v+1] = pw[4+v].y * pe[4+v].y;
      r1[4*v+2] = pw[4+v].z * pe[4+v].z;   r1[4*v+3] = pw[4+v].w * pe[4+v].w;
    }
#pragma unroll
    for (int i = 0; i < 16; ++i) {
      unsigned int dw = f2bf(r0[i]) | (f2bf(r1[i]) << 16);
      *(unsigned int*)(ldsB[buf] + (bc0 + i) * B_STRIDE + k0 * 2) = dw;
    }
  };

  // WMMA compute over one staged K-chunk, B-fragments pipelined one ahead.
  const int arow = wm * 16 + l16;
  auto compute = [&](int buf) {
    const unsigned char* Ab = ldsA[buf];
    const unsigned char* Bb = ldsB[buf];
    Frag32 fa;
    fa.lo = *(const uint4*)(Ab + arow * A_STRIDE +      half * 16);
    fa.hi = *(const uint4*)(Ab + arow * A_STRIDE + 32 + half * 16);
    v16bf a = __builtin_bit_cast(v16bf, fa);

    auto ldb = [&](int j) -> v16bf {
      const int col = wn * 128 + j * 16 + l16;
      Frag32 fb;
      fb.lo = *(const uint4*)(Bb + col * B_STRIDE +      half * 16);
      fb.hi = *(const uint4*)(Bb + col * B_STRIDE + 32 + half * 16);
      return __builtin_bit_cast(v16bf, fb);
    };

    v16bf bcur = ldb(0);
#pragma unroll
    for (int j = 0; j < 8; ++j) {
      v16bf bnext;
      if (j < 7) bnext = ldb(j + 1);   // issue ahead: overlaps with WMMA j
      acc[j] = __builtin_amdgcn_wmma_f32_16x16x32_bf16(
          false, a, false, bcur, (short)0, acc[j], false, false);
      bcur = bnext;
    }
  };

  // ---- prologue: stage chunk 0 ----
  issue_loads(0, 0, 0);
  store_chunk(0);
  __syncthreads();

  // ---- main pipelined loop ----
  for (int kk = 0; kk < NKCH; ++kk) {
    const int cur  = kk & 1;
    const bool more = (kk + 1) < NKCH;
    if (more) {
      const int nk = kk + 1;
      issue_loads(nk / 12, (nk / 4) % 3, (nk & 3) * 32);  // global loads fly
    }
    compute(cur);                      // ~8 WMMAs hide the global latency
    if (more) store_chunk(cur ^ 1);    // waits land here, after compute
    __syncthreads();
  }

  // ---- epilogue: + bias[c]*Berr[b,c], store fp32 NHWC ----
#pragma unroll
  for (int j = 0; j < 8; ++j) {
    const int c  = wn * 128 + j * 16 + l16;
    const float sc = bias[c] * Berr[b * COUTC + c];
#pragma unroll
    for (int i = 0; i < 8; ++i) {
      const int pix = m0 + wm * 16 + half * 8 + i;
      if (pix < NPIX) {
        out[((size_t)b * NPIX + pix) * COUTC + c] = acc[j][i] + sc;
      }
    }
  }
}

extern "C" void kernel_launch(void* const* d_in, const int* in_sizes, int n_in,
                              void* d_out, int out_size, void* d_ws, size_t ws_size,
                              hipStream_t stream) {
  const float* X    = (const float*)d_in[0];  // [64,64,64,128]
  const float* W    = (const float*)d_in[1];  // [3,3,128,256]
  const float* bias = (const float*)d_in[2];  // [256]
  const float* Werr = (const float*)d_in[3];  // [64,3,3,128,256]
  const float* Berr = (const float*)d_in[4];  // [64,256]
  float* out = (float*)d_out;                 // [64,62,62,256]

  dim3 grid(BATCH * NMT);   // 64 samples x 61 M-tiles = 3904 blocks
  dim3 block(256);          // 8 wave32 waves
  conv_aconnect_wmma<<<grid, block, 0, stream>>>(X, W, bias, Werr, Berr, out);
}
